// MultiBandCompressor_85375359910329
// MI455X (gfx1250) — compile-verified, matrix operand order checked by reading, not run on hardware
//
#include <hip/hip_runtime.h>
#include <hip/hip_bf16.h>
#include <math.h>

// ---------------------------------------------------------------------------
// Multi-band compressor for MI455X (gfx1250, wave32).
// STFT(4096, hop 1024) via radix-16 FFT where every 16-point DFT stage is a
// batched complex 16x16x16 matmul on the WMMA pipe (V_WMMA_F32_16X16X4_F32,
// f32 precision). LDS-resident per-frame working set; twiddle table staged
// into LDS with GLOBAL_LOAD_ASYNC_TO_LDS_B128 (ASYNCcnt path); fused
// window/gain math; overlap-add via HW f32 global atomics.
// ---------------------------------------------------------------------------

#define FFT_N   4096
#define HOP     1024
#define KBINS   2049      // FFT_N/2 + 1
#define NFRAMES 509       // (524288 - 4096)/1024 + 1
#define NBC     16        // B*C = 8*2
#define TLEN    524288

typedef float    v2f  __attribute__((ext_vector_type(2)));
typedef float    v8f  __attribute__((ext_vector_type(8)));
typedef _Float16 v16h __attribute__((ext_vector_type(16)));

#if __has_builtin(__builtin_amdgcn_wmma_f32_16x16x4_f32)
#define USE_F32X4 1
#else
#define USE_F32X4 0
#endif

// ---- async global->LDS copy (16B per lane), tracked by ASYNCcnt -----------
__device__ __forceinline__ void async_copy_b128(void* lds_dst, const void* gsrc) {
  // VDST = LDS byte address (low 32 bits of the shared-aperture address),
  // VADDR = 64-bit global address, GV mode (no SADDR).
  unsigned ldsa = (unsigned)(size_t)lds_dst;
  asm volatile("global_load_async_to_lds_b128 %0, %1, off"
               :: "v"(ldsa), "v"(gsrc)
               : "memory");
}

__device__ __forceinline__ void wait_async0() {
#if __has_builtin(__builtin_amdgcn_s_wait_asynccnt)
  __builtin_amdgcn_s_wait_asynccnt(0);
#else
  asm volatile("s_wait_asynccnt 0x0" ::: "memory");
#endif
}

// W16 DFT-matrix fragments (A operand), constant per lane for the whole kernel.
struct W16Frags {
#if USE_F32X4
  v2f wr[4];   // Re(W16)
  v2f wi[4];   // Im(W16)
  v2f wn[4];   // -Im(W16)
#else
  v16h wr, wi, wn;
#endif
};

// base-16 digit reversal of 3 digits (0..4095)
__device__ __forceinline__ int rev16x3(int n) {
  return ((n & 15) << 8) | (n & 240) | (n >> 8);
}

// Build A-matrix fragments for W16[i][a] = exp(-2*pi*I*i*a/16).
// f32 16x16x4 A layout: lanes 0-15 hold M=i; VGPR0 = K (lo half K=4q+0 / hi
// half K=4q+2), VGPR1 = K+1.
__device__ __forceinline__ void make_w16(W16Frags& F, int i16, int hb) {
#if USE_F32X4
#pragma unroll
  for (int q = 0; q < 4; ++q) {
    int k0 = 4 * q + 2 * hb;
    float s0, c0, s1, c1;
    __sincosf(-0.39269908169872414f * (float)((i16 * k0) & 15), &s0, &c0);
    __sincosf(-0.39269908169872414f * (float)((i16 * (k0 + 1)) & 15), &s1, &c1);
    F.wr[q] = (v2f){c0, c1};
    F.wi[q] = (v2f){s0, s1};
    F.wn[q] = (v2f){-s0, -s1};
  }
#else
  // f16 16x16x32 A layout: per lane, halves h<8 cover K = h + 8*hb (K<16),
  // h>=8 cover K>=16 which we zero-pad (only K=0..15 used).
#pragma unroll
  for (int h = 0; h < 16; ++h) {
    float wrv = 0.0f, wiv = 0.0f;
    if (h < 8) {
      int k = h + 8 * hb;
      float s, c;
      __sincosf(-0.39269908169872414f * (float)((i16 * k) & 15), &s, &c);
      wrv = c; wiv = s;
    }
    F.wr[h] = (_Float16)wrv;
    F.wi[h] = (_Float16)wiv;
    F.wn[h] = (_Float16)(-wiv);
  }
#endif
}

// One batch of 16 complex 16-point DFTs: Y(16x16) = W16(16x16) x Z(16x16).
// loadZ(a) returns (Re,Im) of Z[a][c16] for this lane's column (twiddle
// pre-multiplied by the caller). Complex matmul = 4 real f32 matmuls,
// each split into 4 chained K=4 WMMAs.
template <class LF>
__device__ __forceinline__ void dft16_batch(const W16Frags& F, int hb, LF loadZ,
                                            float outr[8], float outi[8]) {
#if USE_F32X4
  v2f bzr[4], bzi[4];
#pragma unroll
  for (int q = 0; q < 4; ++q) {
    int kb = 4 * q + 2 * hb;                 // B layout mirrors A K-split
    float2 z0 = loadZ(kb);
    float2 z1 = loadZ(kb + 1);
    bzr[q] = (v2f){z0.x, z1.x};
    bzi[q] = (v2f){z0.y, z1.y};
  }
  v8f cr = {}; v8f ci = {};
#pragma unroll
  for (int q = 0; q < 4; ++q)
    cr = __builtin_amdgcn_wmma_f32_16x16x4_f32(false, F.wr[q], false, bzr[q], (short)0, cr, false, false);
#pragma unroll
  for (int q = 0; q < 4; ++q)
    cr = __builtin_amdgcn_wmma_f32_16x16x4_f32(false, F.wn[q], false, bzi[q], (short)0, cr, false, false);
#pragma unroll
  for (int q = 0; q < 4; ++q)
    ci = __builtin_amdgcn_wmma_f32_16x16x4_f32(false, F.wr[q], false, bzi[q], (short)0, ci, false, false);
#pragma unroll
  for (int q = 0; q < 4; ++q)
    ci = __builtin_amdgcn_wmma_f32_16x16x4_f32(false, F.wi[q], false, bzr[q], (short)0, ci, false, false);
#else
  // f16 B as filled here: lanes 0-15 hold Z[h][c16] in element h (h=0..15), lanes 16-31
  // are all zero.
  v16h bzr = {}; v16h bzi = {};
  if (hb == 0) {
#pragma unroll
    for (int h = 0; h < 16; ++h) {
      float2 z = loadZ(h);
      bzr[h] = (_Float16)z.x;
      bzi[h] = (_Float16)z.y;
    }
  }
  v8f cr = {}; v8f ci = {};
  cr = __builtin_amdgcn_wmma_f32_16x16x32_f16(false, F.wr, false, bzr, (short)0, cr, false, false);
  cr = __builtin_amdgcn_wmma_f32_16x16x32_f16(false, F.wn, false, bzi, (short)0, cr, false, false);
  ci = __builtin_amdgcn_wmma_f32_16x16x32_f16(false, F.wr, false, bzi, (short)0, ci, false, false);
  ci = __builtin_amdgcn_wmma_f32_16x16x32_f16(false, F.wi, false, bzr, (short)0, ci, false, false);
#endif
#pragma unroll
  for (int v = 0; v < 8; ++v) { outr[v] = cr[v]; outi[v] = ci[v]; }
}

// ---- one batch of each radix-16 stage (straight-line, wave-uniform) -------
// Manual 2x unroll at the call site keeps EXEC untouched around the WMMAs
// (no divergent loop backedge; ISA requires EXEC == all 1s for WMMA).

__device__ __forceinline__ void fft_stage0(float* __restrict__ sre, float* __restrict__ sim,
                                           const W16Frags& F, int c16, int hb, int m) {
  const int base = m << 8;
  float outr[8], outi[8];
  auto loadZ = [&](int a) -> float2 {
    int idx = base + (c16 << 4) + a;
    return make_float2(sre[idx], sim[idx]);
  };
  dft16_batch(F, hb, loadZ, outr, outi);
#pragma unroll
  for (int v = 0; v < 8; ++v) {
    int o = base + (c16 << 4) + v + (hb << 3);
    sre[o] = outr[v]; sim[o] = outi[v];
  }
}

__device__ __forceinline__ void fft_stage1(float* __restrict__ sre, float* __restrict__ sim,
                                           const float2* __restrict__ stw,
                                           const W16Frags& F, int c16, int hb, int m) {
  const int G = m << 8;
  float outr[8], outi[8];
  auto loadZ = [&](int a) -> float2 {   // tw = w_256^{c16*a}
    int idx = G + (a << 4) + c16;
    float xr = sre[idx], xi = sim[idx];
    float2 t = stw[((c16 * a) << 4) & 4095];
    return make_float2(t.x * xr - t.y * xi, t.x * xi + t.y * xr);
  };
  dft16_batch(F, hb, loadZ, outr, outi);
#pragma unroll
  for (int v = 0; v < 8; ++v) {
    int o = G + c16 + ((v + (hb << 3)) << 4);
    sre[o] = outr[v]; sim[o] = outi[v];
  }
}

__device__ __forceinline__ void fft_stage2(float* __restrict__ sre, float* __restrict__ sim,
                                           const float2* __restrict__ stw,
                                           const W16Frags& F, int c16, int hb, int m) {
  const int j = (m << 4) + c16;
  float outr[8], outi[8];
  auto loadZ = [&](int a) -> float2 {   // tw = w_4096^{j*a}
    int idx = (a << 8) + j;
    float xr = sre[idx], xi = sim[idx];
    float2 t = stw[(j * a) & 4095];
    return make_float2(t.x * xr - t.y * xi, t.x * xi + t.y * xr);
  };
  dft16_batch(F, hb, loadZ, outr, outi);
#pragma unroll
  for (int v = 0; v < 8; ++v) {
    int o = j + ((v + (hb << 3)) << 8);
    sre[o] = outr[v]; sim[o] = outi[v];
  }
}

// In-place 4096-pt complex DIT FFT on digit-reversed data in LDS.
// 3 radix-16 stages; each stage = 16 disjoint batches, 2 per wave (8 waves).
__device__ void fft4096_lds(float* __restrict__ sre, float* __restrict__ sim,
                            const float2* __restrict__ stw,
                            const W16Frags& F, int tid) {
  const int lid = tid & 31;
  const int wv  = tid >> 5;     // wave 0..7
  const int c16 = lid & 15;     // column within batch
  const int hb  = lid >> 4;     // half-wave

  fft_stage0(sre, sim, F, c16, hb, wv);
  fft_stage0(sre, sim, F, c16, hb, wv + 8);
  __syncthreads();
  fft_stage1(sre, sim, stw, F, c16, hb, wv);
  fft_stage1(sre, sim, stw, F, c16, hb, wv + 8);
  __syncthreads();
  fft_stage2(sre, sim, stw, F, c16, hb, wv);
  fft_stage2(sre, sim, stw, F, c16, hb, wv + 8);
  __syncthreads();
}

// ---------------------------------------------------------------------------
__global__ void tw_kernel(float2* __restrict__ tw) {
  int j = blockIdx.x * blockDim.x + threadIdx.x;
  if (j < FFT_N) {
    float s, c;
    sincosf(-6.283185307179586f * (float)j * (1.0f / (float)FFT_N), &s, &c);
    tw[j] = make_float2(c, s);
  }
}

__global__ void zero_kernel(float* __restrict__ p, int n) {
  for (int i = blockIdx.x * blockDim.x + threadIdx.x; i < n;
       i += gridDim.x * blockDim.x)
    p[i] = 0.0f;
}

// Forward STFT + per-bin gain-reduction dB. Grid (frame, bc), 256 threads.
__global__ __launch_bounds__(256) void fwd_kernel(
    const float* __restrict__ audio, const float* __restrict__ win,
    const float* __restrict__ thr, const float* __restrict__ ratio,
    const float* __restrict__ knee, const float2* __restrict__ twg,
    float* __restrict__ Xr, float* __restrict__ Xi, float* __restrict__ Gr) {
  __shared__ float  sre[FFT_N];
  __shared__ float  sim[FFT_N];
  __shared__ float2 stw[FFT_N];
  const int tid = threadIdx.x;
  const int f = blockIdx.x, bc = blockIdx.y;
  const size_t abase = (size_t)bc * TLEN + (size_t)f * HOP;

  // Kick off the 32 KB twiddle-table copy on the async pipe (no VGPR round
  // trip, tracked by ASYNCcnt) while we window + digit-reverse the frame.
  for (int t = tid * 2; t < FFT_N; t += 512)
    async_copy_b128(&stw[t], &twg[t]);

  W16Frags F;
  make_w16(F, tid & 15, (tid & 31) >> 4);

  for (int t = tid; t < FFT_N; t += 256) {
    float x = audio[abase + t] * win[t];
    int r = rev16x3(t);              // digit-reversed DIT input
    sre[r] = x;
    sim[r] = 0.0f;
  }
  __builtin_prefetch(&audio[abase + FFT_N], 0, 1);  // global_prefetch_b8
  wait_async0();
  __syncthreads();

  fft4096_lds(sre, sim, stw, F, tid);

  const size_t obase = ((size_t)f * NBC + bc) * KBINS;
  for (int k = tid; k < KBINS; k += 256) {
    float xr = sre[k], xi = sim[k];
    float mag = sqrtf(xr * xr + xi * xi);
    float mdb = 20.0f * log10f(fmaxf(mag, 1e-8f));
    float th = thr[k], rt = ratio[k], kw = knee[k];
    float ks = th - 0.5f * kw, ke = th + 0.5f * kw;
    float over = mdb - th;
    float slope = 1.0f - 1.0f / rt;
    float hard = (mdb < th) ? 0.0f : over * slope;
    float kf = fminf(fmaxf((mdb - ks) / kw, 0.0f), 1.0f);
    float kg = kf * kf * over * slope;
    float gr = (mdb >= ks && mdb <= ke) ? kg : hard;
    Xr[obase + k] = xr;
    Xi[obase + k] = xi;
    Gr[obase + k] = gr;
  }
}

// Sequential attack/release envelope per bin; converts gr_db -> linear gain
// g = clip(env,1e-8) * 10^(makeup/20)  (since db2lin(-sm) == clip(env,1e-8)).
__global__ void smooth_kernel(float* __restrict__ Gr, const float* __restrict__ mk) {
  int b = blockIdx.x * blockDim.x + threadIdx.x;
  if (b >= NBC * KBINS) return;
  int bc = b / KBINS;
  int k  = b - bc * KBINS;
  const float acoef = 1.0f - __expf(-1.0f / 441.0f);   // 10 ms @ 44.1 kHz
  const float rcoef = 1.0f - __expf(-1.0f / 4410.0f);  // 100 ms
  const float L2_10_20 = 0.16609640474436813f;         // log2(10)/20
  float mkg = exp2f(mk[k] * L2_10_20);
  float env = 0.0f;
  for (int f = 0; f < NFRAMES; ++f) {
    size_t idx = ((size_t)f * NBC + bc) * KBINS + k;
    float gr  = Gr[idx];
    float tgt = exp2f(-gr * L2_10_20);                 // 10^(-gr/20)
    float cf  = (tgt < env) ? acoef : rcoef;
    env = env + cf * (tgt - env);
    Gr[idx] = fmaxf(env, 1e-8f) * mkg;
  }
}

// Inverse: Hermitian-extend gained spectrum, irfft = real(FFT(conj(X)))/N,
// windowed overlap-add via HW f32 atomics.
__global__ __launch_bounds__(256) void inv_kernel(
    const float* __restrict__ Xr, const float* __restrict__ Xi,
    const float* __restrict__ Gn, const float* __restrict__ win,
    const float2* __restrict__ twg, float* __restrict__ out) {
  __shared__ float  sre[FFT_N];
  __shared__ float  sim[FFT_N];
  __shared__ float2 stw[FFT_N];
  const int tid = threadIdx.x;
  const int f = blockIdx.x, bc = blockIdx.y;

  for (int t = tid * 2; t < FFT_N; t += 512)
    async_copy_b128(&stw[t], &twg[t]);

  W16Frags F;
  make_w16(F, tid & 15, (tid & 31) >> 4);

  const size_t obase = ((size_t)f * NBC + bc) * KBINS;
  for (int t = tid; t < KBINS; t += 256) {
    float g  = Gn[obase + t];
    float vr = Xr[obase + t] * g;
    float vi = Xi[obase + t] * g;
    int r1 = rev16x3(t);
    sre[r1] = vr; sim[r1] = -vi;          // conj for inverse trick
    if (t != 0 && t != 2048) {
      int r2 = rev16x3(FFT_N - t);        // conj(Hermitian mirror)
      sre[r2] = vr; sim[r2] = vi;
    }
  }
  wait_async0();
  __syncthreads();

  fft4096_lds(sre, sim, stw, F, tid);

  const size_t wb = (size_t)bc * TLEN + (size_t)f * HOP;
  const float inv_n = 1.0f / (float)FFT_N;
  for (int n = tid; n < FFT_N; n += 256) {
    float y = sre[n] * inv_n * win[n];
    __hip_atomic_fetch_add(&out[wb + n], y, __ATOMIC_RELAXED,
                           __HIP_MEMORY_SCOPE_AGENT);
  }
}

// ---------------------------------------------------------------------------
extern "C" void kernel_launch(void* const* d_in, const int* in_sizes, int n_in,
                              void* d_out, int out_size, void* d_ws, size_t ws_size,
                              hipStream_t stream) {
  (void)in_sizes; (void)n_in; (void)ws_size;
  const float* audio = (const float*)d_in[0];
  const float* win   = (const float*)d_in[1];
  const float* thr   = (const float*)d_in[2];
  const float* ratio = (const float*)d_in[3];
  const float* mk    = (const float*)d_in[4];
  const float* knee  = (const float*)d_in[5];
  float* out = (float*)d_out;

  char* ws = (char*)d_ws;
  float2* tw = (float2*)ws;                       // 32 KB twiddle table
  size_t off = 32768;
  const size_t spec = (size_t)NFRAMES * NBC * KBINS;  // 16,687,056 elems
  float* Xr = (float*)(ws + off); off += spec * sizeof(float);
  float* Xi = (float*)(ws + off); off += spec * sizeof(float);
  float* Gr = (float*)(ws + off);

  tw_kernel<<<(FFT_N + 255) / 256, 256, 0, stream>>>(tw);
  zero_kernel<<<2048, 256, 0, stream>>>(out, out_size);
  fwd_kernel<<<dim3(NFRAMES, NBC), 256, 0, stream>>>(audio, win, thr, ratio,
                                                     knee, tw, Xr, Xi, Gr);
  smooth_kernel<<<(NBC * KBINS + 255) / 256, 256, 0, stream>>>(Gr, mk);
  inv_kernel<<<dim3(NFRAMES, NBC), 256, 0, stream>>>(Xr, Xi, Gr, win, tw, out);
}
